// StaticRecurrentLayer_backup_22814866276391
// MI455X (gfx1250) — matrix operand order repaired and checked by reading, not compiled, not run on hardware
//
#include <hip/hip_runtime.h>
#include <hip/hip_bf16.h>

// ---------------------------------------------------------------------------
// MNN recurrent layer on gfx1250: per step, dual bf16 WMMA GEMMs
//   mean = u_del @ W^T ,  var = (s_del^2) @ (W*W)^T
// fused with the moment-activation epilogue and Euler state update.
// ---------------------------------------------------------------------------

#define N_NEU   2048
#define BATCH   256
#define TSTEPS  100
#define DELAYP1 6          // u_del(t) = u_{t-6}

typedef __bf16 bf16_t;
typedef __attribute__((ext_vector_type(16))) __bf16 bf16x16;
typedef __attribute__((ext_vector_type(8)))  __bf16 bf16x8;
typedef __attribute__((ext_vector_type(8)))  float  f32x8;

static __device__ __forceinline__ f32x8 wmma_bf16(bf16x16 a, bf16x16 b, f32x8 c) {
  // D = A(16x32 bf16) x B(32x16 bf16) + C(16x16 f32)   -> v_wmma_f32_16x16x32_bf16
  return __builtin_amdgcn_wmma_f32_16x16x32_bf16(false, a, false, b, (short)0, c,
                                                 false, false);
}

// A-matrix (16-bit, 16x32) per-lane fragment, contiguous K order:
//   lane<16 : K = {k0..k0+15}; lane>=16: K = {k0+16..k0+31}.
// p already points at (row, k0 + half*16): fragment = p[0..15].
static __device__ __forceinline__ bf16x16 loadA(const bf16_t* p) {
  return *(const bf16x16*)p;
}

// ---------------------------------------------------------------------------
__global__ __launch_bounds__(256)
void mnn_init(const float* __restrict__ W,
              bf16_t* __restrict__ Wbf, bf16_t* __restrict__ W2bf,
              float* __restrict__ uSt, float* __restrict__ sSt,
              bf16_t* __restrict__ uR, bf16_t* __restrict__ s2R) {
  const size_t NN = (size_t)N_NEU * N_NEU;
  const size_t BN = (size_t)BATCH * N_NEU;
  size_t i = (size_t)blockIdx.x * blockDim.x + threadIdx.x;
  size_t stride = (size_t)gridDim.x * blockDim.x;
  for (size_t k = i; k < NN; k += stride) {
    float w = W[k];
    Wbf[k]  = (bf16_t)w;
    W2bf[k] = (bf16_t)(w * w);
  }
  for (size_t k = i; k < BN; k += stride) { uSt[k] = 0.0f; sSt[k] = 0.0f; }
  for (size_t k = i; k < 8 * BN; k += stride) {
    uR[k]  = (bf16_t)0.0f;
    s2R[k] = (bf16_t)0.0f;
  }
}

// ---------------------------------------------------------------------------
// One time step. Block = 128 threads (4 waves); wave tile = 32x32 of C for
// BOTH GEMMs (2x2 WMMA tiles x 2 matrices = 8 accumulators). Block covers a
// 64(batch) x 64(neuron) region; grid = (2048/64, 256/64) = (32, 4).
__global__ __launch_bounds__(128)
void mnn_step(const bf16_t* __restrict__ Wbf,  const bf16_t* __restrict__ W2bf,
              const bf16_t* __restrict__ uDel, const bf16_t* __restrict__ s2Del,
              bf16_t* __restrict__ uNext,      bf16_t* __restrict__ s2Next,
              const float* __restrict__ ff_mean, const float* __restrict__ ff_std,
              float* __restrict__ uSt, float* __restrict__ sSt,
              float* __restrict__ Uout, float* __restrict__ Sout, int t) {
  const int lane = threadIdx.x & 31;
  const int wave = threadIdx.x >> 5;
  const int half = lane >> 4;      // 0: lanes 0-15, 1: lanes 16-31
  const int l    = lane & 15;

  const int m0 = blockIdx.y * 64 + (wave & 1) * 32;   // batch-tile origin
  const int n0 = blockIdx.x * 64 + (wave >> 1) * 32;  // neuron-tile origin

  const f32x8 zf = (f32x8)0.0f;
  f32x8 accU[2][2] = {{zf, zf}, {zf, zf}};
  f32x8 accS[2][2] = {{zf, zf}, {zf, zf}};

  const size_t aRow0 = (size_t)(m0 + l)      * N_NEU;   // A rows (mi = 0/1)
  const size_t aRow1 = (size_t)(m0 + 16 + l) * N_NEU;
  const size_t bRow0 = (size_t)(n0 + l)      * N_NEU;   // B rows (ni = 0/1)
  const size_t bRow1 = (size_t)(n0 + 16 + l) * N_NEU;

  for (int k0 = 0; k0 < N_NEU; k0 += 32) {
    const int ka = k0 + half * 16;   // A fragment base: 16 contiguous K
    const int kb = k0 + half * 16;   // B fragment base: 16 contiguous K

    if (k0 + 96 < N_NEU) {           // keep W rows streaming out of L2
      __builtin_prefetch(Wbf  + bRow0 + k0 + 96, 0, 0);
      __builtin_prefetch(W2bf + bRow1 + k0 + 96, 0, 0);
    }

    bf16x16 au0 = loadA(uDel  + aRow0 + ka);
    bf16x16 au1 = loadA(uDel  + aRow1 + ka);
    bf16x16 as0 = loadA(s2Del + aRow0 + ka);
    bf16x16 as1 = loadA(s2Del + aRow1 + ka);

    bf16x16 bw0 = *(const bf16x16*)(Wbf  + bRow0 + kb);
    bf16x16 bw1 = *(const bf16x16*)(Wbf  + bRow1 + kb);
    bf16x16 b20 = *(const bf16x16*)(W2bf + bRow0 + kb);
    bf16x16 b21 = *(const bf16x16*)(W2bf + bRow1 + kb);

    accU[0][0] = wmma_bf16(au0, bw0, accU[0][0]);
    accU[0][1] = wmma_bf16(au0, bw1, accU[0][1]);
    accU[1][0] = wmma_bf16(au1, bw0, accU[1][0]);
    accU[1][1] = wmma_bf16(au1, bw1, accU[1][1]);
    accS[0][0] = wmma_bf16(as0, b20, accS[0][0]);
    accS[0][1] = wmma_bf16(as0, b21, accS[0][1]);
    accS[1][0] = wmma_bf16(as1, b20, accS[1][0]);
    accS[1][1] = wmma_bf16(as1, b21, accS[1][1]);
  }

  // Epilogue: C/D layout -> element r is (M = r + 8*half, N = l).
  #pragma unroll
  for (int mi = 0; mi < 2; ++mi) {
    #pragma unroll
    for (int ni = 0; ni < 2; ++ni) {
      #pragma unroll
      for (int r = 0; r < 8; ++r) {
        const int bb = m0 + mi * 16 + half * 8 + r;
        const int nn = n0 + ni * 16 + l;
        const size_t idx = (size_t)bb * N_NEU + nn;

        float mc = accU[mi][ni][r] + 0.5f + ff_mean[idx];   // + BG_INPUT + ff
        float fs = ff_std[idx];
        float var = accS[mi][ni][r] + fs * fs;
        float sc = sqrtf(fmaxf(var, 0.0f));

        float x = (mc - 20.0f) / (sc + 1e-6f);              // (mean-VTH)/(std+EPS)
        float drive = (x > 0.0f) ? (x + log1pf(expf(-x))) : log1pf(expf(x));
        float rate = drive / (1.0f + 5.0f * drive);         // TREF = 5
        float fano = 1.0f / (1.0f + drive);
        float ostd = sqrtf(rate * fano + 1e-6f);

        float uo = uSt[idx], so = sSt[idx];
        float un = 0.9f * uo + 0.1f * rate;                 // a = DT/TAU = 0.1
        float sn = 0.9f * so + 0.1f * ostd;

        Uout[idx * TSTEPS + t] = uo;   // outputs record PRE-update state
        Sout[idx * TSTEPS + t] = so;
        uSt[idx] = un;
        sSt[idx] = sn;
        uNext[idx]  = (bf16_t)un;      // push u_{t+1} into ring slot (t+1)&7
        s2Next[idx] = (bf16_t)(sn * sn);
      }
    }
  }
}

// ---------------------------------------------------------------------------
extern "C" void kernel_launch(void* const* d_in, const int* in_sizes, int n_in,
                              void* d_out, int out_size, void* d_ws, size_t ws_size,
                              hipStream_t stream) {
  const float* ff_mean = (const float*)d_in[0];
  const float* ff_std  = (const float*)d_in[1];
  const float* W       = (const float*)d_in[2];

  const size_t NN = (size_t)N_NEU * N_NEU;
  const size_t BN = (size_t)BATCH * N_NEU;

  float* Uout = (float*)d_out;
  float* Sout = Uout + BN * TSTEPS;

  // Workspace layout (~36 MB): bf16 W, bf16 W^2, f32 u/s state, 8-slot rings.
  bf16_t* Wbf  = (bf16_t*)d_ws;
  bf16_t* W2bf = Wbf + NN;
  float*  uSt  = (float*)(W2bf + NN);
  float*  sSt  = uSt + BN;
  bf16_t* uR   = (bf16_t*)(sSt + BN);
  bf16_t* s2R  = uR + 8 * BN;

  mnn_init<<<1024, 256, 0, stream>>>(W, Wbf, W2bf, uSt, sSt, uR, s2R);

  dim3 grid(N_NEU / 64, BATCH / 64);   // (32, 4)
  for (int t = 0; t < TSTEPS; ++t) {
    // ring[k & 7] holds u_k; read u_{t-6} from slot (t+2)&7, write u_{t+1}
    // to slot (t+1)&7 (disjoint, so no intra-kernel hazard).
    const bf16_t* uDel   = uR  + (size_t)((t + 2) & 7) * BN;
    const bf16_t* s2Del  = s2R + (size_t)((t + 2) & 7) * BN;
    bf16_t*       uNext  = uR  + (size_t)((t + 1) & 7) * BN;
    bf16_t*       s2Next = s2R + (size_t)((t + 1) & 7) * BN;
    mnn_step<<<grid, 128, 0, stream>>>(Wbf, W2bf, uDel, s2Del, uNext, s2Next,
                                       ff_mean, ff_std, uSt, sSt, Uout, Sout, t);
  }
}
